// PointAttentionNetwork_47433618817640
// MI455X (gfx1250) — matrix operand order repaired and checked by reading, not compiled, not run on hardware
//
#include <hip/hip_runtime.h>
#include <cstdint>
#include <cstddef>

// ---------------------------------------------------------------------------
// PointAttentionNetwork fused implementation for gfx1250 (MI455X), wave32.
//   stage 0: prep       - fold BN into scale/bias, convert W to bf16
//   stage 1: conv GEMMs - v_wmma_f32_16x16x32_bf16, x tile staged via LDS
//   stage 2: flash attn - online softmax; K tiles via async global->LDS
//                         (ASYNCcnt), V tiles via TDM tensor_load_to_lds
//                         (TENSORcnt), double-buffered, WMMA consumes from LDS
// ---------------------------------------------------------------------------

typedef __attribute__((ext_vector_type(16))) __bf16 v16bf;
typedef __attribute__((ext_vector_type(8)))  __bf16 v8bf;
typedef __attribute__((ext_vector_type(8)))  float  v8f;
typedef __attribute__((ext_vector_type(4)))  int    v4i;
typedef __attribute__((ext_vector_type(8)))  int    v8i;
typedef __attribute__((ext_vector_type(4)))  unsigned v4u;
typedef __attribute__((address_space(1)))    v4i    v4i_g;   // global AS element
typedef __attribute__((address_space(3)))    v4i    v4i_l;   // LDS AS element

constexpr int   B_  = 8;
constexpr int   C_  = 512;
constexpr int   N_  = 4096;
constexpr int   Cr_ = 128;   // C/4
constexpr int   Ch_ = 256;   // C/2
constexpr float EPS_ = 1e-5f;

#define DEVINL static __device__ __forceinline__

// ---- async global->LDS copy of 16 bytes per lane --------------------------
#if __has_builtin(__builtin_amdgcn_global_load_async_to_lds_b128)
DEVINL void async_cp16(const __bf16* g, __bf16* l) {
  __builtin_amdgcn_global_load_async_to_lds_b128((v4i_g*)g, (v4i_l*)l, 0, 0);
}
#else
DEVINL void async_cp16(const __bf16* g, __bf16* l) {
  asm volatile("global_load_async_to_lds_b128 %0, %1, off"
               :: "v"((unsigned)(uintptr_t)l),
                  "v"((unsigned long long)(uintptr_t)g)
               : "memory");
}
#endif

#if __has_builtin(__builtin_amdgcn_s_wait_asynccnt)
#define WAIT_ASYNC(n) __builtin_amdgcn_s_wait_asynccnt(n)
#else
#define WAIT_ASYNC(n) asm volatile("s_wait_asynccnt %0" :: "i"(n) : "memory")
#endif

// ---- TDM: one-instruction 2D tile DMA global->LDS (TENSORcnt) -------------
#if __has_builtin(__builtin_amdgcn_tensor_load_to_lds)
#define HAVE_TDM 1
#if __has_builtin(__builtin_amdgcn_s_wait_tensorcnt)
#define WAIT_TENSOR(n) __builtin_amdgcn_s_wait_tensorcnt(n)
#else
#define WAIT_TENSOR(n) asm volatile("s_wait_tensorcnt %0" :: "i"(n) : "memory")
#endif
// D# per CDNA5 ISA: groups 0/1 bit-packed below.
// This toolchain's builtin takes 6 args (extra int32x8 before cpol).
DEVINL void tdm_load_2d(const void* gptr, void* lptr,
                        unsigned tile_d0, unsigned tile_d1,
                        unsigned tens_d0, unsigned tens_d1,
                        unsigned long long stride0_elems) {
  const unsigned long long ga = (unsigned long long)(uintptr_t)gptr;
  v4u g0;
  g0[0] = 1u;                                   // count=1 (valid), load mode
  g0[1] = (unsigned)(uintptr_t)lptr;            // lds_addr (bytes)
  g0[2] = (unsigned)ga;                         // global_addr[31:0]
  g0[3] = (unsigned)(ga >> 32) | (2u << 30);    // global_addr[56:32] | type=2
  v8i g1;
  g1[0] = (int)(1u << 16);                      // data_size=1 -> 2-byte elems
  g1[1] = (int)((tens_d0 & 0xFFFFu) << 16);                         // dim0 lo
  g1[2] = (int)((tens_d0 >> 16) | ((tens_d1 & 0xFFFFu) << 16));     // dim0 hi | dim1 lo
  g1[3] = (int)((tens_d1 >> 16) | (tile_d0 << 16));                 // dim1 hi | tile0
  g1[4] = (int)(tile_d1 & 0xFFFFu);                                 // tile1 (tile2=0)
  g1[5] = (int)(unsigned)stride0_elems;                             // stride0[31:0]
  g1[6] = (int)(unsigned)((stride0_elems >> 32) & 0xFFFFu);         // stride0[47:32]
  g1[7] = 0;                                                        // stride1=0
  const v4i z4 = {0, 0, 0, 0};
  const v8i z8 = {0, 0, 0, 0, 0, 0, 0, 0};
  __builtin_amdgcn_tensor_load_to_lds(g0, g1, z4, z4, z8, 0);
}
#else
#define HAVE_TDM 0
#endif

DEVINL v16bf cat16(v8bf a, v8bf b) {
  return __builtin_shufflevector(a, b, 0,1,2,3,4,5,6,7,8,9,10,11,12,13,14,15);
}

// ---------------------------------------------------------------------------
// Stage 0: fold BN, convert weights to bf16
// ---------------------------------------------------------------------------
__global__ void prep_kernel(
    const float* __restrict__ W1, const float* __restrict__ W2, const float* __restrict__ W3,
    const float* g1, const float* be1, const float* mu1, const float* va1,
    const float* g2, const float* be2, const float* mu2, const float* va2,
    const float* g3, const float* be3, const float* mu3, const float* va3,
    __bf16* Wb1, __bf16* Wb2, __bf16* Wb3,
    float* s1, float* o1, float* s2, float* o2, float* s3, float* o3)
{
  const int idx = blockIdx.x * blockDim.x + threadIdx.x;
  const int n1 = Cr_ * C_;          // 65536
  const int n3 = Ch_ * C_;          // 131072
  if (idx < n1)                 Wb1[idx]          = (__bf16)W1[idx];
  else if (idx < 2 * n1)        Wb2[idx - n1]     = (__bf16)W2[idx - n1];
  else if (idx < 2 * n1 + n3)   Wb3[idx - 2 * n1] = (__bf16)W3[idx - 2 * n1];
  if (idx < Cr_) {
    float sc = g1[idx] * rsqrtf(va1[idx] + EPS_);
    s1[idx] = sc; o1[idx] = be1[idx] - mu1[idx] * sc;
    sc = g2[idx] * rsqrtf(va2[idx] + EPS_);
    s2[idx] = sc; o2[idx] = be2[idx] - mu2[idx] * sc;
  }
  if (idx < Ch_) {
    float sc = g3[idx] * rsqrtf(va3[idx] + EPS_);
    s3[idx] = sc; o3[idx] = be3[idx] - mu3[idx] * sc;
  }
}

// ---------------------------------------------------------------------------
// Stage 1: 1x1 conv + BN + ReLU as WMMA GEMM.
//   TRANS=1 : out[b, n, o]  (point-major,   for aT / kT)
//   TRANS=0 : out[b, o, n]  (channel-major, for vC)
// grid = (N/16, Cout/128, B), block = 256 (8 waves; wave -> one 16-row o tile)
// ---------------------------------------------------------------------------
template <int TRANS>
__global__ __launch_bounds__(256) void conv_bn_relu(
    const float* __restrict__ x,       // [B, C, N]
    const __bf16* __restrict__ Wb,     // [Cout, C]
    const float* __restrict__ scale, const float* __restrict__ bias,
    __bf16* __restrict__ out, int Cout)
{
  __shared__ __bf16 xbuf[16][32];      // [n][c] tile, 1 KB

  const int lane = threadIdx.x & 31;
  const int wave = threadIdx.x >> 5;
  const int hi   = lane >> 4;          // 0 / 1 (half-group)
  const int l16  = lane & 15;
  const int b    = blockIdx.z;
  const int nb   = blockIdx.x * 16;
  const int ob   = (blockIdx.y * 8 + wave) * 16;

  const float* xb = x + (size_t)b * C_ * N_;

  v8f acc = {};
  for (int cc = 0; cc < C_; cc += 32) {
    {
      int idx = threadIdx.x;
#pragma unroll
      for (int r = 0; r < 2; ++r, idx += 256) {
        const int c = idx >> 4, n = idx & 15;
        xbuf[n][c] = (__bf16)xb[(size_t)(cc + c) * N_ + nb + n];
      }
    }
    __syncthreads();
    const __bf16* wrow = Wb + (size_t)(ob + l16) * C_ + cc + hi * 16;
    v16bf afrag = cat16(*(const v8bf*)wrow, *(const v8bf*)(wrow + 8));
    const __bf16* xr = &xbuf[l16][hi * 16];
    v16bf bfrag = cat16(*(const v8bf*)xr, *(const v8bf*)(xr + 8));
    acc = __builtin_amdgcn_wmma_f32_16x16x32_bf16(false, afrag, false, bfrag,
                                                  (short)0, acc, false, false);
    __syncthreads();
  }

  const int orow0 = ob + hi * 8;       // C-layout: row = j + 8*hi, col = l16
  if (TRANS) {
    v8bf res;
#pragma unroll
    for (int j = 0; j < 8; ++j) {
      float v = acc[j] * scale[orow0 + j] + bias[orow0 + j];
      res[j] = (__bf16)fmaxf(v, 0.0f);
    }
    *(v8bf*)(out + ((size_t)b * N_ + nb + l16) * Cout + orow0) = res;
  } else {
#pragma unroll
    for (int j = 0; j < 8; ++j) {
      float v = acc[j] * scale[orow0 + j] + bias[orow0 + j];
      out[((size_t)b * Cout + orow0 + j) * N_ + nb + l16] = (__bf16)fmaxf(v, 0.0f);
    }
  }
}

// ---------------------------------------------------------------------------
// Stage 2: flash attention with async + TDM double buffering.
// grid = (N/64, B), block = 128 (4 waves; wave -> 16 queries, full Ch=256)
// ---------------------------------------------------------------------------
#if HAVE_TDM
constexpr int NISSUE = 4;    // K-only async b128 per wave per key block
#else
constexpr int NISSUE = 12;   // K + V async b128 per wave per key block
#endif

__global__ __launch_bounds__(128) void flash_attn(
    const __bf16* __restrict__ aT,   // [B, N, Cr]
    const __bf16* __restrict__ kT,   // [B, N, Cr]
    const __bf16* __restrict__ vC,   // [B, Ch, N]
    float* __restrict__ out)         // [B, Ch, N]
{
  __shared__ __bf16 kbuf[2][32][128];   // K tiles, 2 x 8 KB   [key][c]
  __shared__ __bf16 vbuf[2][256][32];   // V tiles, 2 x 16 KB  [vc][key]
  __shared__ __bf16 p_lds[4][16][32];   // per-wave P staging (C-layout -> A-layout)

  const int lane = threadIdx.x & 31;
  const int wave = threadIdx.x >> 5;
  const int hi   = lane >> 4;
  const int l16  = lane & 15;
  const int b    = blockIdx.y;
  const int qb   = blockIdx.x * 64 + wave * 16;

  const __bf16* aTb = aT + (size_t)b * N_ * Cr_;
  const __bf16* kTb = kT + (size_t)b * N_ * Cr_;
  const __bf16* vCb = vC + (size_t)b * Ch_ * N_;

  // issue one key block's copies: K via async b128 (contiguous 8 KB region),
  // V via one TDM 2D tile DMA (32 x 256 elems, row stride N) issued by wave 0.
  auto issue = [&](int buf, int kb) {
    const __bf16* gk = kTb + (size_t)kb * Cr_;
#pragma unroll
    for (int j = 0; j < 4; ++j) {
      const int off = (((wave * 4 + j) * 32) + lane) * 8;   // bf16 elements
      async_cp16(gk + off, &kbuf[buf][0][0] + off);
    }
#if HAVE_TDM
    if (wave == 0)
      tdm_load_2d(vCb + kb, &vbuf[buf][0][0],
                  /*tile*/ 32, Ch_, /*tensor*/ N_, Ch_, /*stride*/ N_);
#else
#pragma unroll
    for (int j = 0; j < 8; ++j) {
      const int instr = wave * 8 + j;          // 0..31
      const int vc    = instr * 8 + (lane >> 2);
      const int part  = lane & 3;
      async_cp16(vCb + (size_t)vc * N_ + kb + part * 8, &vbuf[buf][vc][part * 8]);
    }
#endif
  };

  // Q A-fragments: 4 chunks of K=32 over Cr=128 (per-wave, stays in VGPRs)
  v16bf qf[4];
#pragma unroll
  for (int c4 = 0; c4 < 4; ++c4) {
    const __bf16* qrow = aTb + (size_t)(qb + l16) * Cr_ + c4 * 32 + hi * 16;
    qf[c4] = cat16(*(const v8bf*)qrow, *(const v8bf*)(qrow + 8));
  }

  v8f O[16] = {};                          // 16 q x 256 vc accumulators
  float m[8], l[8];
#pragma unroll
  for (int j = 0; j < 8; ++j) { m[j] = -1.0e30f; l[j] = 0.0f; }

  issue(0, 0);                             // prologue prefetch

  for (int kb = 0; kb < N_; kb += 32) {
    const int cur = (kb >> 5) & 1;
    if (kb + 32 < N_) {
      issue(cur ^ 1, kb + 32);
      WAIT_ASYNC(NISSUE);
#if HAVE_TDM
      if (wave == 0) WAIT_TENSOR(1);
#endif
    } else {
      WAIT_ASYNC(0);
#if HAVE_TDM
      if (wave == 0) WAIT_TENSOR(0);
#endif
    }
    __syncthreads();                       // current buffer ready block-wide

    // ---- S = Q . K^T for 32 keys; pipelined K fragment loads from LDS ----
    v8f S0 = {}, S1 = {};
    {
      const __bf16* k0p = &kbuf[cur][l16][hi * 16];
      const __bf16* k1p = &kbuf[cur][16 + l16][hi * 16];
      v16bf kf0 = cat16(*(const v8bf*)k0p, *(const v8bf*)(k0p + 8));
      v16bf kf1 = cat16(*(const v8bf*)k1p, *(const v8bf*)(k1p + 8));
#pragma unroll
      for (int c4 = 0; c4 < 4; ++c4) {
        v16bf nf0, nf1;
        if (c4 < 3) {                      // prefetch next chunk before WMMA
          nf0 = cat16(*(const v8bf*)(k0p + (c4 + 1) * 32),
                      *(const v8bf*)(k0p + (c4 + 1) * 32 + 8));
          nf1 = cat16(*(const v8bf*)(k1p + (c4 + 1) * 32),
                      *(const v8bf*)(k1p + (c4 + 1) * 32 + 8));
        }
        S0 = __builtin_amdgcn_wmma_f32_16x16x32_bf16(false, qf[c4], false, kf0,
                                                     (short)0, S0, false, false);
        S1 = __builtin_amdgcn_wmma_f32_16x16x32_bf16(false, qf[c4], false, kf1,
                                                     (short)0, S1, false, false);
        kf0 = nf0; kf1 = nf1;
      }
    }

    // ---- online softmax; batched butterfly reductions over 16-lane groups ----
    float rm[8], al[8], rs[8];
#pragma unroll
    for (int j = 0; j < 8; ++j) rm[j] = fmaxf(S0[j], S1[j]);
#pragma unroll
    for (int st = 8; st >= 1; st >>= 1) {   // 8 independent bpermutes per step
      float t[8];
#pragma unroll
      for (int j = 0; j < 8; ++j) t[j] = __shfl_xor(rm[j], st);
#pragma unroll
      for (int j = 0; j < 8; ++j) rm[j] = fmaxf(rm[j], t[j]);
    }
#pragma unroll
    for (int j = 0; j < 8; ++j) {
      const float mn = fmaxf(m[j], rm[j]);
      al[j] = __expf(m[j] - mn);
      S0[j] = __expf(S0[j] - mn);
      S1[j] = __expf(S1[j] - mn);
      rs[j] = S0[j] + S1[j];
      m[j] = mn;
    }
#pragma unroll
    for (int st = 8; st >= 1; st >>= 1) {
      float t[8];
#pragma unroll
      for (int j = 0; j < 8; ++j) t[j] = __shfl_xor(rs[j], st);
#pragma unroll
      for (int j = 0; j < 8; ++j) rs[j] += t[j];
    }
#pragma unroll
    for (int j = 0; j < 8; ++j) {
      l[j] = l[j] * al[j] + rs[j];
      const int qr = j + 8 * hi;
      p_lds[wave][qr][l16]      = (__bf16)S0[j];
      p_lds[wave][qr][16 + l16] = (__bf16)S1[j];
    }
#pragma unroll
    for (int t = 0; t < 16; ++t)
#pragma unroll
      for (int j = 0; j < 8; ++j) O[t][j] *= al[j];

    // P A-fragment from LDS (same-wave LDS ops are in-order -> RAW safe)
    const __bf16* prow = &p_lds[wave][l16][hi * 16];
    v16bf pf = cat16(*(const v8bf*)prow, *(const v8bf*)(prow + 8));

    // ---- O += P . V; pipelined V fragment loads from LDS ----
    {
      const __bf16* vbase = &vbuf[cur][l16][hi * 16];   // + t*16 rows of 32
      v16bf vf = cat16(*(const v8bf*)vbase, *(const v8bf*)(vbase + 8));
#pragma unroll
      for (int t = 0; t < 16; ++t) {
        v16bf nvf;
        if (t < 15) {
          const __bf16* nv = vbase + (size_t)(t + 1) * 16 * 32;
          nvf = cat16(*(const v8bf*)nv, *(const v8bf*)(nv + 8));
        }
        O[t] = __builtin_amdgcn_wmma_f32_16x16x32_bf16(false, pf, false, vf,
                                                       (short)0, O[t], false, false);
        vf = nvf;
      }
    }
    __syncthreads();   // all waves done reading buf[cur] before it is refilled
  }

  // ---- epilogue: out[b, vc, q] = v + O / l ----
#pragma unroll
  for (int j = 0; j < 8; ++j) l[j] = 1.0f / l[j];
  float* outb = out + (size_t)b * Ch_ * N_;
#pragma unroll
  for (int t = 0; t < 16; ++t) {
    const int vc = t * 16 + l16;
    const size_t base = (size_t)vc * N_ + qb + 8 * hi;   // q = qb + j + 8*hi
    v8bf vres = *(const v8bf*)(vCb + base);
#pragma unroll
    for (int j = 0; j < 8; ++j)
      outb[base + j] = (float)vres[j] + O[t][j] * l[j];
  }
}

// ---------------------------------------------------------------------------
// host launcher
// ---------------------------------------------------------------------------
extern "C" void kernel_launch(void* const* d_in, const int* in_sizes, int n_in,
                              void* d_out, int out_size, void* d_ws, size_t ws_size,
                              hipStream_t stream) {
  (void)in_sizes; (void)n_in; (void)out_size; (void)ws_size;

  const float* x  = (const float*)d_in[0];
  const float* W1 = (const float*)d_in[1];
  const float* g1 = (const float*)d_in[2];
  const float* b1 = (const float*)d_in[3];
  const float* m1 = (const float*)d_in[4];
  const float* v1 = (const float*)d_in[5];
  const float* W2 = (const float*)d_in[6];
  const float* g2 = (const float*)d_in[7];
  const float* b2 = (const float*)d_in[8];
  const float* m2 = (const float*)d_in[9];
  const float* v2 = (const float*)d_in[10];
  const float* W3 = (const float*)d_in[11];
  const float* g3 = (const float*)d_in[12];
  const float* b3 = (const float*)d_in[13];
  const float* m3 = (const float*)d_in[14];
  const float* v3 = (const float*)d_in[15];

  uint8_t* ws = (uint8_t*)d_ws;
  // workspace layout (~34 MB)
  __bf16* Wb1 = (__bf16*)(ws + 0);                   // 128*512*2 = 128 KB
  __bf16* Wb2 = (__bf16*)(ws + 131072);              // 128 KB
  __bf16* Wb3 = (__bf16*)(ws + 262144);              // 256 KB
  float*  s1  = (float*)(ws + 524288);
  float*  o1  = (float*)(ws + 524800);
  float*  s2  = (float*)(ws + 525312);
  float*  o2  = (float*)(ws + 525824);
  float*  s3  = (float*)(ws + 526336);
  float*  o3  = (float*)(ws + 527360);
  __bf16* aT  = (__bf16*)(ws + (1u << 20));          // [B,N,Cr]  8 MB
  __bf16* kT  = (__bf16*)(ws + 9437184u);            // [B,N,Cr]  8 MB
  __bf16* vCp = (__bf16*)(ws + 17825792u);           // [B,Ch,N] 16 MB

  prep_kernel<<<1024, 256, 0, stream>>>(W1, W2, W3,
      g1, b1, m1, v1, g2, b2, m2, v2, g3, b3, m3, v3,
      Wb1, Wb2, Wb3, s1, o1, s2, o2, s3, o3);

  dim3 gc12(N_ / 16, Cr_ / 128, B_);                 // (256, 1, 8)
  conv_bn_relu<1><<<gc12, 256, 0, stream>>>(x, Wb1, s1, o1, aT, Cr_);
  conv_bn_relu<1><<<gc12, 256, 0, stream>>>(x, Wb2, s2, o2, kT, Cr_);
  dim3 gc3(N_ / 16, Ch_ / 128, B_);                  // (256, 2, 8)
  conv_bn_relu<0><<<gc3, 256, 0, stream>>>(x, Wb3, s3, o3, vCp, Ch_);

  dim3 gf(N_ / 64, B_);                              // (64, 8)
  flash_attn<<<gf, 128, 0, stream>>>(aT, kT, vCp, (float*)d_out);
}
